// QRnn_64656437674395
// MI455X (gfx1250) — matrix operand order repaired and checked by reading, not compiled, not run on hardware
//
#include <hip/hip_runtime.h>
#include <hip/hip_bf16.h>

typedef __attribute__((ext_vector_type(16))) _Float16 v16h;
typedef __attribute__((ext_vector_type(8)))  float    v8f;

__device__ __forceinline__ float fast_sigmoid(float x) {
    return 1.0f / (1.0f + __expf(-x));
}
__device__ __forceinline__ float fast_tanh(float x) {
    return 1.0f - 2.0f / (1.0f + __expf(2.0f * x));
}

// ---------------------------------------------------------------------------
// GEMM: D(MxN) = A(f16, MxKp row-major) * Wt(f16, N x Kp == W^T) + bias
// Each wave: one 16-wide N tile x TWO 16-row M tiles (2 wmma / K-step).
// A tiles (2 x 16x32 f16) staged via async global->LDS, shared by all waves.
// mode: 1 = qrnn act (tanh | sigmoid), store f16
//       2 = relu, store f16
//       3 = relu, store f32 (final layer -> d_out)
// ---------------------------------------------------------------------------
union ABld { float4 f4[2]; v16h v; };

__global__ void qrnn_gemm_wmma_f16(const _Float16* __restrict__ A,
                                   const _Float16* __restrict__ Wt,
                                   const float* __restrict__ bias,
                                   void* __restrict__ Dv,
                                   int Kp, int N, int ldD, int Nout, int mode) {
    __shared__ __align__(16) _Float16 ldsA[2 * 16 * 40];  // 80B row stride

    const int lane  = threadIdx.x & 31;
    const int wave  = threadIdx.x >> 5;
    const int wpb   = blockDim.x >> 5;
    const int mtile = blockIdx.x;                  // covers 32 rows
    const int ntile = blockIdx.y * wpb + wave;
    const int row   = lane & 15;                   // A row in tile / D col in tile
    const int khalf = lane >> 4;

    const _Float16* Abase = A  + (size_t)mtile * 32 * Kp;
    const _Float16* Bcol  = Wt + (size_t)(ntile * 16 + row) * Kp;

    v8f c0 = {}, c1 = {};
    for (int k0 = 0; k0 < Kp; k0 += 32) {
        __syncthreads();   // previous iteration's LDS reads complete
        // async copy of two 16x32 f16 A tiles into LDS (128 x 16B chunks)
        for (int ch = threadIdx.x; ch < 128; ch += blockDim.x) {
            const int tr  = ch >> 2;               // 0..31: row across both tiles
            const int seg = ch & 3;
            const unsigned ldsOff =
                (unsigned)(size_t)&ldsA[tr * 40 + seg * 8];     // low 32b = LDS offset
            const unsigned long long gaddr =
                (unsigned long long)(size_t)(Abase + (size_t)tr * Kp + k0 + seg * 8);
            asm volatile("global_load_async_to_lds_b128 %0, %1, off"
                         :: "v"(ldsOff), "v"(gaddr) : "memory");
        }
        asm volatile("s_wait_asynccnt 0" ::: "memory");
        __syncthreads();   // tiles visible to all waves

        if (k0 + 32 < Kp) __builtin_prefetch(Bcol + k0 + 32, 0, 0);

        ABld a0, a1, b;
        // A layout: v[0..7] = K(khalf*8 ..), v[8..15] = K(16+khalf*8 ..)
        a0.f4[0] = *(const float4*)(&ldsA[row * 40 + khalf * 8]);
        a0.f4[1] = *(const float4*)(&ldsA[row * 40 + 16 + khalf * 8]);
        a1.f4[0] = *(const float4*)(&ldsA[(16 + row) * 40 + khalf * 8]);
        a1.f4[1] = *(const float4*)(&ldsA[(16 + row) * 40 + 16 + khalf * 8]);
        // B layout: v[0..7] = K(khalf*8 ..), v[8..15] = K(16+khalf*8 ..) for column (lane&15)
        b.f4[0] = *(const float4*)(Bcol + k0 + khalf * 8);
        b.f4[1] = *(const float4*)(Bcol + k0 + 16 + khalf * 8);

        c0 = __builtin_amdgcn_wmma_f32_16x16x32_f16(false, a0.v, false, b.v,
                                                    (short)0, c0, false, false);
        c1 = __builtin_amdgcn_wmma_f32_16x16x32_f16(false, a1.v, false, b.v,
                                                    (short)0, c1, false, false);
    }

    // D layout: VGPR r -> tile row r (lanes 0-15) / r+8 (lanes 16-31), col = lane&15
    const int gcol = ntile * 16 + row;
    if (gcol < Nout) {
        const float bv = bias[gcol];
        const int   H  = N >> 1;
        const int   r0 = mtile * 32 + (khalf ? 8 : 0);
#pragma unroll
        for (int t = 0; t < 2; ++t) {
            const v8f& c = t ? c1 : c0;
            const int rbase = r0 + t * 16;
#pragma unroll
            for (int r = 0; r < 8; ++r) {
                float v = c[r] + bv;
                if (mode == 1)      v = (gcol < H) ? fast_tanh(v) : fast_sigmoid(v);
                else                v = fmaxf(v, 0.0f);
                const size_t idx = (size_t)(rbase + r) * ldD + gcol;
                if (mode == 3) ((float*)Dv)[idx]    = v;
                else           ((_Float16*)Dv)[idx] = (_Float16)v;
            }
        }
    }
}

// ---------------------------------------------------------------------------
// Vanilla tanh RNN, hidden=4. One workgroup, thread t = (b=t/4, h=t&3),
// shared h-state in LDS, sequential over S. Writes f16 act (B*S, 32-padded).
// ---------------------------------------------------------------------------
__global__ void rnn_tanh_kernel(const float* __restrict__ x,
                                const float* __restrict__ wih,
                                const float* __restrict__ whh,
                                const float* __restrict__ bih,
                                const float* __restrict__ bhh,
                                _Float16* __restrict__ act, int S) {
    __shared__ float hbuf[256];
    const int t = threadIdx.x;      // 256 = 64 batches * 4 hidden
    const int b = t >> 2, h = t & 3;
    const float wi = wih[h];
    const float bb = bih[h] + bhh[h];
    const float w0 = whh[h * 4 + 0], w1 = whh[h * 4 + 1];
    const float w2 = whh[h * 4 + 2], w3 = whh[h * 4 + 3];
    hbuf[t] = 0.0f;
    __syncthreads();
    for (int s = 0; s < S; ++s) {
        float pre = x[(size_t)b * S + s] * wi + bb;
        float hn  = fast_tanh(pre + hbuf[b * 4 + 0] * w0 + hbuf[b * 4 + 1] * w1 +
                                    hbuf[b * 4 + 2] * w2 + hbuf[b * 4 + 3] * w3);
        __syncthreads();
        hbuf[t] = hn;
        __syncthreads();
        act[((size_t)b * S + s) * 32 + h] = (_Float16)hn;
    }
}

// fo-pool: h_t = f_t*h_{t-1} + (1-f_t)*z_t   (z/f pre-activated f16)
__global__ void qrnn_scan_kernel(const _Float16* __restrict__ zf,
                                 _Float16* __restrict__ act,
                                 int Bn, int S, int H, int Kp) {
    int t = blockIdx.x * blockDim.x + threadIdx.x;
    if (t >= Bn * H) return;
    int b = t / H, h = t % H;
    const _Float16* zrow = zf + (size_t)b * S * (2 * H);
    _Float16* arow = act + (size_t)b * S * Kp;
    float hs = 0.0f;
    for (int s = 0; s < S; ++s) {
        float z = (float)zrow[(size_t)s * 2 * H + h];
        float f = (float)zrow[(size_t)s * 2 * H + H + h];
        hs = f * hs + (1.0f - f) * z;
        arow[(size_t)s * Kp + h] = (_Float16)hs;
    }
}

// mean over channel axis: (B,S,C) f16 -> pooled (B, Sp) f16, zero-padded S..Sp
union P8 { float4 f; _Float16 h[8]; };
__global__ void pool_mean_kernel(const _Float16* __restrict__ act,
                                 _Float16* __restrict__ pooled,
                                 int Bn, int S, int C, int Sp) {
    int t = blockIdx.x * blockDim.x + threadIdx.x;
    if (t >= Bn * Sp) return;
    int b = t / Sp, sp = t % Sp;
    float v = 0.0f;
    if (sp < S) {
        const float4* p = (const float4*)(act + ((size_t)b * S + sp) * C);
        float sum = 0.0f;
        for (int c = 0; c < C / 8; ++c) {
            P8 u; u.f = p[c];
#pragma unroll
            for (int j = 0; j < 8; ++j) sum += (float)u.h[j];
        }
        v = sum / (float)C;
    }
    pooled[(size_t)b * Sp + sp] = (_Float16)v;
}

// W (f32, K x N) -> Wt (f16, Np x Kp), transposed + zero padded
__global__ void wconv_kernel(const float* __restrict__ W, _Float16* __restrict__ Wt,
                             int K, int N, int Kp, int Np) {
    int t = blockIdx.x * blockDim.x + threadIdx.x;
    if (t >= Np * Kp) return;
    int n = t / Kp, k = t % Kp;
    float v = (n < N && k < K) ? W[(size_t)k * N + n] : 0.0f;
    Wt[t] = (_Float16)v;
}

__global__ void zero_f16_kernel(_Float16* __restrict__ p, size_t n) {
    size_t t = (size_t)blockIdx.x * blockDim.x + threadIdx.x;
    for (; t < n; t += (size_t)gridDim.x * blockDim.x) p[t] = (_Float16)0.0f;
}

// ---------------------------------------------------------------------------
extern "C" void kernel_launch(void* const* d_in, const int* in_sizes, int n_in,
                              void* d_out, int out_size, void* d_ws, size_t ws_size,
                              hipStream_t stream) {
    (void)in_sizes; (void)n_in; (void)out_size; (void)ws_size;
    const int Bn = 64, S = 3000, M = Bn * S;  // 192000 rows (multiple of 32)

    const float* x   = (const float*)d_in[0];
    const float* wih = (const float*)d_in[1];
    const float* whh = (const float*)d_in[2];
    const float* bih = (const float*)d_in[3];
    const float* bhh = (const float*)d_in[4];
    const float* qw[5] = {(const float*)d_in[5],  (const float*)d_in[7],
                          (const float*)d_in[9],  (const float*)d_in[11],
                          (const float*)d_in[13]};
    const float* qb[5] = {(const float*)d_in[6],  (const float*)d_in[8],
                          (const float*)d_in[10], (const float*)d_in[12],
                          (const float*)d_in[14]};
    const float* w1 = (const float*)d_in[15]; const float* b1 = (const float*)d_in[16];
    const float* w2 = (const float*)d_in[17]; const float* b2 = (const float*)d_in[18];
    const float* w3 = (const float*)d_in[19]; const float* b3 = (const float*)d_in[20];

    // ---- workspace carving (256B aligned) ----
    char* ws = (char*)d_ws;
    _Float16*  ZF     = (_Float16*) (ws);                 // 192000*512*2 = 196,608,000
    _Float16*  ACTA   = (_Float16*) (ws + 196608000);     // 192000*256*2 =  98,304,000
    _Float16*  ACTB   = (_Float16*) (ws + 294912000);     //                 98,304,000
    _Float16*  WT     = (_Float16*) (ws + 393216000);     // max 1024*3008*2 = 6,160,384
    char*      smallw = ws + 399376640;
    _Float16*  pooled = (_Float16*) (smallw);             // 64*3008*2 = 385,024
    _Float16*  d1h    = (_Float16*) (smallw + 385024);    // 64*1024*2 = 131,072
    _Float16*  d2h    = (_Float16*) (smallw + 516096);    // 64*128*2  =  16,384

    auto gemm = [&](const _Float16* A, int Mrows, int Kp, int N, const float* bias,
                    void* D, int ldD, int Nout, int mode) {
        int ntiles = N / 16;
        int wpb = ntiles < 8 ? ntiles : 8;
        dim3 grid(Mrows / 32, ntiles / wpb);
        qrnn_gemm_wmma_f16<<<grid, wpb * 32, 0, stream>>>(A, WT, bias, D, Kp, N,
                                                          ldD, Nout, mode);
    };
    auto wconv = [&](const float* W, int K, int N, int Kp, int Np) {
        int n = Np * Kp;
        wconv_kernel<<<(n + 255) / 256, 256, 0, stream>>>(W, WT, K, N, Kp, Np);
    };
    auto zfill = [&](_Float16* p, size_t n) {
        zero_f16_kernel<<<2048, 256, 0, stream>>>(p, n);
    };

    // ---- stage 1: tanh RNN (hidden 4) -> ACTA f16 (192000 x 32-pad) ----
    zfill(ACTA, (size_t)M * 32);
    rnn_tanh_kernel<<<1, 256, 0, stream>>>(x, wih, whh, bih, bhh, ACTA, S);

    // ---- stage 2: five QRNN layers (WMMA GEMM + fused act, then fo-pool scan)
    const int Cin[5]  = {4, 16, 32, 64, 128};
    const int Kpin[5] = {32, 32, 32, 64, 128};
    const int Hout[5] = {16, 32, 64, 128, 256};
    _Float16* actIn = ACTA;
    _Float16* actOut = ACTB;
    for (int i = 0; i < 5; ++i) {
        const int H = Hout[i], N = 2 * H;
        const int KpOut = (H < 32) ? 32 : H;
        wconv(qw[i], Cin[i], N, Kpin[i], N);
        gemm(actIn, M, Kpin[i], N, qb[i], ZF, N, N, /*mode=*/1);
        if (KpOut != H) zfill(actOut, (size_t)M * KpOut);
        int nt = Bn * H;
        qrnn_scan_kernel<<<(nt + 255) / 256, 256, 0, stream>>>(ZF, actOut, Bn, S, H, KpOut);
        _Float16* tmp = actIn; actIn = actOut; actOut = tmp;
    }

    // ---- stage 3: channel-mean pool -> pooled (64 x 3008-pad f16) ----
    {
        int n = Bn * 3008;
        pool_mean_kernel<<<(n + 255) / 256, 256, 0, stream>>>(actIn, pooled, Bn, S, 256, 3008);
    }

    // ---- stage 4: MLP (relu fused in epilogue, f16 intermediates) ----
    wconv(w1, 3000, 1024, 3008, 1024);
    gemm(pooled, 64, 3008, 1024, b1, d1h, 1024, 1024, /*relu f16*/2);

    wconv(w2, 1024, 128, 1024, 128);
    gemm(d1h, 64, 1024, 128, b2, d2h, 128, 128, /*relu f16*/2);

    wconv(w3, 128, 10, 128, 16);  // pad N 10 -> 16, zero rows
    gemm(d2h, 64, 128, 16, b3, d_out, /*ldD=*/10, /*Nout=*/10, /*relu f32*/3);
}
